// MFF_81398220194078
// MI455X (gfx1250) — compile-verified, matrix operand order checked by reading, not run on hardware
//
#include <hip/hip_runtime.h>
#include <math.h>

// ---------------------------------------------------------------------------
// MI455X (gfx1250) fused KBA block.
// Heavy path: per-pixel att(32) x kb_w(32x1728) bilinear contraction fused with
// the unfold-dot, run on v_wmma_f32_16x16x32_bf16 (f32 accumulate).
// B matrices (kb_w slices / 1x1 weights) are staged in LDS once per block and
// read back as ds_load_b128 fragments, eliminating redundant per-wave L2 reads.
// ---------------------------------------------------------------------------

typedef __bf16 bf16_t;
typedef __attribute__((ext_vector_type(16))) __bf16 v16bf;
typedef __attribute__((ext_vector_type(8)))  float  v8f;

#define HWPIX 16384   // 128*128
#define BATCH 2

__device__ __forceinline__ bf16_t f2bf(float f) {
  union { float f; unsigned u; } v; v.f = f;
  unsigned r = v.u + 0x7FFFu + ((v.u >> 16) & 1u);   // round-to-nearest-even
  unsigned short s = (unsigned short)(r >> 16);
  bf16_t o; __builtin_memcpy(&o, &s, 2); return o;
}

// Load a wave32 16x32 bf16 fragment (works for both A rows and B columns,
// and for both global and LDS-resident source after inlining).
// ISA 7.12.2: lanes 0-15 hold K=0..7 & 16..23, lanes 16-31 hold K=8..15 & 24..31.
__device__ __forceinline__ v16bf load_frag16(const bf16_t* base, int stride, int kOff) {
  const int lane = threadIdx.x & 31;
  const int rc   = lane & 15;
  const int hi   = lane >> 4;
  const bf16_t* p = base + (size_t)rc * stride + kOff + hi * 8;
  v16bf f;
#pragma unroll
  for (int t = 0; t < 8; ++t) { f[t] = p[t]; f[8 + t] = p[16 + t]; }
  return f;
}

__device__ __forceinline__ v8f wmma_bf16(v16bf a, v16bf b, v8f c) {
  return __builtin_amdgcn_wmma_f32_16x16x32_bf16(false, a, false, b, (short)0, c,
                                                 false, false);
}

// --------------------------- prep kernels ----------------------------------

// x (B,48,H,W) f32 -> xT (B*HW, 64) bf16, K padded 48->64 with zeros.
__global__ __launch_bounds__(256) void k_prep_x(const float* __restrict__ x,
                                                bf16_t* __restrict__ xT) {
  int n = blockIdx.x * 256 + threadIdx.x;              // < 32768
  int b = n >> 14, np = n & 16383;
  bf16_t* dst = xT + (size_t)n * 64;
  const float* src = x + ((size_t)(b * 48) << 14) + np;
#pragma unroll
  for (int c = 0; c < 48; ++c) dst[c] = f2bf(src[(size_t)c << 14]);
#pragma unroll
  for (int c = 48; c < 64; ++c) dst[c] = f2bf(0.f);
}

// Pack all weight matrices into WMMA-friendly bf16 layouts.
__global__ __launch_bounds__(256) void k_prep_w(
    const float* __restrict__ w1a, const float* __restrict__ w1c,
    const float* __restrict__ w211, const float* __restrict__ kbw,
    const float* __restrict__ kbb, const float* __restrict__ wproj,
    bf16_t* __restrict__ wcat, bf16_t* __restrict__ bfKW,
    bf16_t* __restrict__ bfBias, bf16_t* __restrict__ wProjB) {
  int i = blockIdx.x * 256 + threadIdx.x;
  if (i < 224 * 64) {                       // concatenated 1x1 weights, K pad 64
    int c = i >> 6, k = i & 63;
    float v = 0.f;
    if (k < 48) {
      if (c < 96)        v = w1a[c * 48 + k];
      else if (c < 192)  v = w1c[(c - 96) * 48 + k];
      else               v = w211[(c - 192) * 48 + k];
    }
    wcat[i] = f2bf(v);
    return;
  }
  i -= 224 * 64;
  if (i < 18 * 96 * 32) {                   // kb_w permuted: [(j*96+ch)*32 + s]
    int s = i & 31, r = i >> 5;
    int ch = r % 96, j = r / 96;
    bfKW[i] = f2bf(kbw[s * 1728 + (ch >> 1) * 36 + (ch & 1) * 18 + j]);
    return;
  }
  i -= 18 * 96 * 32;
  if (i < 96 * 32) {                        // kb_b: [ch*32 + s]
    int s = i & 31, ch = i >> 5;
    bfBias[i] = f2bf(kbb[s * 96 + ch]);
    return;
  }
  i -= 96 * 32;
  if (i < 48 * 96) { wProjB[i] = f2bf(wproj[i]); return; }   // [co*96 + k]
}

// Channel means of x for SCA branch. One block per (b,c).
__global__ __launch_bounds__(256) void k_mean(const float* __restrict__ x,
                                              float* __restrict__ means) {
  __shared__ float red[256];
  int bc = blockIdx.x;                                  // b*48 + c
  const float* p = x + ((size_t)bc << 14);
  float s = 0.f;
  for (int i = threadIdx.x; i < HWPIX; i += 256) s += p[i];
  red[threadIdx.x] = s;
  __syncthreads();
  for (int st = 128; st > 0; st >>= 1) {
    if (threadIdx.x < st) red[threadIdx.x] += red[threadIdx.x + st];
    __syncthreads();
  }
  if (threadIdx.x == 0) means[bc] = red[0] * (1.f / HWPIX);
}

__global__ void k_sca(const float* __restrict__ means, const float* __restrict__ wsca,
                      const float* __restrict__ bsca, float* __restrict__ sca) {
  int t = blockIdx.x * blockDim.x + threadIdx.x;
  if (t >= 192) return;
  int b = t / 96, co = t % 96;
  float s = bsca[co];
  for (int ci = 0; ci < 48; ++ci) s += wsca[co * 48 + ci] * means[b * 48 + ci];
  sca[t] = s;
}

// --------------------------- GEMM: three 1x1 convs at once -----------------
// (B*HW,64) x (64,224) -> hidA(96) | hidC(96) | attLin(32), channel-major f32.
// Weight matrix (28 KB) staged in LDS once per block.
__global__ __launch_bounds__(256) void k_gemm1(
    const bf16_t* __restrict__ xT, const bf16_t* __restrict__ wcat,
    const float* __restrict__ b1a, const float* __restrict__ b1c,
    const float* __restrict__ b211, float* __restrict__ hidA,
    float* __restrict__ hidC, float* __restrict__ attL) {
  __shared__ __align__(16) bf16_t sW[224 * 64];
  for (int i = threadIdx.x; i < (224 * 64) / 8; i += 256)      // 1792 x uint4
    ((uint4*)sW)[i] = ((const uint4*)wcat)[i];

  int wave = blockIdx.x * 8 + (threadIdx.x >> 5);
  int n0 = wave << 4;                                   // 16 pixels / wave
  int b = n0 >> 14, np = n0 & 16383;
  int lane = threadIdx.x & 31, col = lane & 15, hi = lane >> 4;
  v16bf a0 = load_frag16(xT + (size_t)n0 * 64, 64, 0);
  v16bf a1 = load_frag16(xT + (size_t)n0 * 64, 64, 32);
  __syncthreads();

  for (int nt = 0; nt < 14; ++nt) {
    v8f d = {0.f, 0.f, 0.f, 0.f, 0.f, 0.f, 0.f, 0.f};
    d = wmma_bf16(a0, load_frag16(sW + (size_t)nt * 16 * 64, 64, 0), d);
    d = wmma_bf16(a1, load_frag16(sW + (size_t)nt * 16 * 64, 64, 32), d);
    int c = nt * 16 + col;
    float bias; float* dst;
    if (c < 96)       { bias = b1a[c];        dst = hidA + ((size_t)(b * 96 + c) << 14); }
    else if (c < 192) { bias = b1c[c - 96];   dst = hidC + ((size_t)(b * 96 + c - 96) << 14); }
    else              { bias = b211[c - 192]; dst = attL + ((size_t)(b * 32 + c - 192) << 14); }
    dst += np + hi * 8;
#pragma unroll
    for (int vg = 0; vg < 8; ++vg) dst[vg] = d[vg] + bias;
  }
}

// --------------------------- depthwise 3x3 ---------------------------------
__global__ __launch_bounds__(256) void k_dw(const float* __restrict__ src,
                                            const float* __restrict__ w,
                                            const float* __restrict__ bias,
                                            float* __restrict__ dst) {
  int idx = blockIdx.x * 256 + threadIdx.x;             // < 2*96*HW
  int np = idx & 16383;
  int t = idx >> 14;                                    // b*96 + c
  int c = t % 96;
  int y = np >> 7, xx = np & 127;
  const float* p = src + ((size_t)t << 14);
  const float* wc = w + c * 9;
  float acc = bias[c];
#pragma unroll
  for (int ky = 0; ky < 3; ++ky)
#pragma unroll
    for (int kx = 0; kx < 3; ++kx) {
      int yy = y + ky - 1, xv = xx + kx - 1;
      float v = ((unsigned)yy < 128u && (unsigned)xv < 128u) ? p[yy * 128 + xv] : 0.f;
      acc += v * wc[ky * 3 + kx];
    }
  dst[idx] = acc;
}

// --------------------------- attention branch ------------------------------
// grouped conv (24 out, 2-in groups) -> SimpleGate -> 1x1(12->32)*gamma + attLin
__global__ __launch_bounds__(256) void k_att(
    const float* __restrict__ x, const float* __restrict__ wc2a,
    const float* __restrict__ bc2a, const float* __restrict__ wc2b,
    const float* __restrict__ bc2b, const float* __restrict__ attg,
    const float* __restrict__ attL, bf16_t* __restrict__ attT) {
  int n = blockIdx.x * 256 + threadIdx.x;               // < 32768
  int b = n >> 14, np = n & 16383;
  int y = np >> 7, xx = np & 127;
  float t[24];
#pragma unroll
  for (int co = 0; co < 24; ++co) {
    float acc = bc2a[co];
#pragma unroll
    for (int ci = 0; ci < 2; ++ci) {
      const float* p = x + ((size_t)(b * 48 + co * 2 + ci) << 14);
      const float* wp = wc2a + (co * 2 + ci) * 9;
#pragma unroll
      for (int ky = 0; ky < 3; ++ky)
#pragma unroll
        for (int kx = 0; kx < 3; ++kx) {
          int yy = y + ky - 1, xv = xx + kx - 1;
          float v = ((unsigned)yy < 128u && (unsigned)xv < 128u) ? p[yy * 128 + xv] : 0.f;
          acc += v * wp[ky * 3 + kx];
        }
    }
    t[co] = acc;
  }
  float g[12];
#pragma unroll
  for (int i = 0; i < 12; ++i) g[i] = t[i] * t[i + 12];
  bf16_t* dst = attT + (size_t)n * 32;
#pragma unroll
  for (int s = 0; s < 32; ++s) {
    float a = bc2b[s];
#pragma unroll
    for (int i = 0; i < 12; ++i) a += wc2b[s * 12 + i] * g[i];
    a = a * attg[s] + attL[((size_t)(b * 32 + s) << 14) + np];
    dst[s] = f2bf(a);
  }
}

// --------------------------- fused KBA core --------------------------------
// Per wave: 16 pixels. 6 channel-tiles x (1 bias WMMA + 18 j-WMMAs); each WMMA
// tile multiplied elementwise by the register-cached uf patch. The 18 KB kb_w
// slice + 1 KB kb_b slice for the current channel tile are staged in LDS by
// the whole block (8 waves share them). Epilogue fuses *ga1 + uf residual,
// exact GELU(x1), *sca, bf16 store into the projection GEMM's A layout.
__global__ __launch_bounds__(256) void k_kba(
    const bf16_t* __restrict__ attT, const bf16_t* __restrict__ bfKW,
    const bf16_t* __restrict__ bfBias, const float* __restrict__ uf,
    const float* __restrict__ x1, const float* __restrict__ sca,
    const float* __restrict__ ga1, bf16_t* __restrict__ yT) {
  __shared__ __align__(16) bf16_t sKW[18 * 16 * 32];    // [(j*16+chLocal)*32+s]
  __shared__ __align__(16) bf16_t sBias[16 * 32];

  int wave = blockIdx.x * 8 + (threadIdx.x >> 5);
  int n0 = wave << 4;                                   // 16 pixels, same row
  int b = n0 >> 14, np = n0 & 16383;
  int y = np >> 7, x0 = np & 127;
  int lane = threadIdx.x & 31, col = lane & 15, hi = lane >> 4;

  v16bf afrag = load_frag16(attT + (size_t)n0 * 32, 32, 0);  // att A-fragment
  int xb = x0 + hi * 8;                                 // this lane's pixel base

  for (int ct = 0; ct < 6; ++ct) {
    // ---- stage kb_w / kb_b slice for this channel tile into LDS ----
    __syncthreads();
    for (int i = threadIdx.x; i < 1152; i += 256) {     // 18 j-rows x 64 uint4
      int j = i >> 6, o = i & 63;
      ((uint4*)sKW)[i] =
          ((const uint4*)(bfKW + (size_t)(j * 96 + ct * 16) * 32))[o];
    }
    if (threadIdx.x < 64)
      ((uint4*)sBias)[threadIdx.x] =
          ((const uint4*)(bfBias + (size_t)(ct * 16) * 32))[threadIdx.x];
    __syncthreads();

    int ch = ct * 16 + col;                             // this lane's out channel
    int g2 = ch & ~1;                                   // group's first in-channel
    // Register-cached 3x3 patches: 2 in-channels x 3 rows x 10 x-positions.
    float patch[6][10];
#pragma unroll
    for (int cc = 0; cc < 2; ++cc)
#pragma unroll
      for (int ky = 0; ky < 3; ++ky) {
        int yy = y + ky - 1;
        bool yok = (unsigned)yy < 128u;
        const float* row = uf + ((size_t)(b * 96 + g2 + cc) << 14) + yy * 128;
#pragma unroll
        for (int ix = 0; ix < 10; ++ix) {
          int xv = xb - 1 + ix;
          patch[cc * 3 + ky][ix] = (yok && (unsigned)xv < 128u) ? row[xv] : 0.f;
        }
      }

    v8f acc = {0.f, 0.f, 0.f, 0.f, 0.f, 0.f, 0.f, 0.f};
    acc = wmma_bf16(afrag, load_frag16(sBias, 32, 0), acc);

#pragma unroll
    for (int cc = 0; cc < 2; ++cc)
#pragma unroll
      for (int ky = 0; ky < 3; ++ky)
#pragma unroll
        for (int kx = 0; kx < 3; ++kx) {
          int j = cc * 9 + ky * 3 + kx;
          v8f d = {0.f, 0.f, 0.f, 0.f, 0.f, 0.f, 0.f, 0.f};
          d = wmma_bf16(afrag, load_frag16(sKW + (size_t)(j * 16) * 32, 32, 0), d);
#pragma unroll
          for (int vg = 0; vg < 8; ++vg)
            acc[vg] += d[vg] * patch[cc * 3 + ky][vg + kx];
        }

    float gav  = ga1[ch];
    float scav = sca[b * 96 + ch];
    const float* x1p = x1 + ((size_t)(b * 96 + ch) << 14) + np + hi * 8;
    bf16_t* yp = yT + (size_t)(n0 + hi * 8) * 96 + ch;
#pragma unroll
    for (int vg = 0; vg < 8; ++vg) {
      float ufres = (ch & 1) ? patch[4][vg + 1] : patch[1][vg + 1];   // center tap
      float x2 = acc[vg] * gav + ufres;
      float xv = x1p[vg];
      float ge = 0.5f * xv * (1.f + erff(xv * 0.70710678118f));       // exact GELU
      yp[(size_t)vg * 96] = f2bf(ge * x2 * scav);
    }
  }
}

// --------------------------- projection GEMM 96->48 ------------------------
__global__ __launch_bounds__(256) void k_proj(
    const bf16_t* __restrict__ yT, const bf16_t* __restrict__ wProjB,
    const float* __restrict__ bproj, float* __restrict__ out) {
  __shared__ __align__(16) bf16_t sW[48 * 96];
  for (int i = threadIdx.x; i < (48 * 96) / 8; i += 256)       // 576 x uint4
    ((uint4*)sW)[i] = ((const uint4*)wProjB)[i];

  int wave = blockIdx.x * 8 + (threadIdx.x >> 5);
  int n0 = wave << 4;
  int b = n0 >> 14, np = n0 & 16383;
  int lane = threadIdx.x & 31, col = lane & 15, hi = lane >> 4;
  v16bf a0 = load_frag16(yT + (size_t)n0 * 96, 96, 0);
  v16bf a1 = load_frag16(yT + (size_t)n0 * 96, 96, 32);
  v16bf a2 = load_frag16(yT + (size_t)n0 * 96, 96, 64);
  __syncthreads();

  for (int nt = 0; nt < 3; ++nt) {
    v8f d = {0.f, 0.f, 0.f, 0.f, 0.f, 0.f, 0.f, 0.f};
    d = wmma_bf16(a0, load_frag16(sW + (size_t)nt * 16 * 96, 96, 0), d);
    d = wmma_bf16(a1, load_frag16(sW + (size_t)nt * 16 * 96, 96, 32), d);
    d = wmma_bf16(a2, load_frag16(sW + (size_t)nt * 16 * 96, 96, 64), d);
    int c = nt * 16 + col;
    float* dst = out + ((size_t)(b * 48 + c) << 14) + np + hi * 8;
    float bias = bproj[c];
#pragma unroll
    for (int vg = 0; vg < 8; ++vg) dst[vg] = d[vg] + bias;
  }
}

// --------------------------- host launcher ---------------------------------
extern "C" void kernel_launch(void* const* d_in, const int* in_sizes, int n_in,
                              void* d_out, int out_size, void* d_ws, size_t ws_size,
                              hipStream_t stream) {
  (void)in_sizes; (void)n_in; (void)out_size; (void)ws_size;
  const float* x     = (const float*)d_in[0];
  const float* w1a   = (const float*)d_in[1];
  const float* b1a   = (const float*)d_in[2];
  const float* wdwa  = (const float*)d_in[3];
  const float* bdwa  = (const float*)d_in[4];
  const float* w1c   = (const float*)d_in[5];
  const float* b1c   = (const float*)d_in[6];
  const float* wdwc  = (const float*)d_in[7];
  const float* bdwc  = (const float*)d_in[8];
  const float* wsca  = (const float*)d_in[9];
  const float* bsca  = (const float*)d_in[10];
  const float* wc2a  = (const float*)d_in[11];
  const float* bc2a  = (const float*)d_in[12];
  const float* wc2b  = (const float*)d_in[13];
  const float* bc2b  = (const float*)d_in[14];
  const float* w211  = (const float*)d_in[15];
  const float* b211  = (const float*)d_in[16];
  const float* wproj = (const float*)d_in[17];
  const float* bproj = (const float*)d_in[18];
  const float* kbw   = (const float*)d_in[19];
  const float* kbb   = (const float*)d_in[20];
  const float* attg  = (const float*)d_in[21];
  const float* ga1   = (const float*)d_in[22];

  char* ws = (char*)d_ws;
  size_t off = 0;
  auto take = [&](size_t bytes) -> char* {
    char* p = ws + off;
    off += (bytes + 255) & ~(size_t)255;
    return p;
  };
  bf16_t* xT     = (bf16_t*)take((size_t)BATCH * HWPIX * 64 * 2);
  bf16_t* wcat   = (bf16_t*)take(224 * 64 * 2);
  bf16_t* bfKW   = (bf16_t*)take((size_t)18 * 96 * 32 * 2);
  bf16_t* bfBias = (bf16_t*)take(96 * 32 * 2);
  bf16_t* wProjB = (bf16_t*)take(48 * 96 * 2);
  float*  means  = (float*)take(96 * 4);
  float*  sca    = (float*)take(192 * 4);
  float*  hidA   = (float*)take((size_t)BATCH * 96 * HWPIX * 4);
  float*  hidC   = (float*)take((size_t)BATCH * 96 * HWPIX * 4);
  float*  x1     = (float*)take((size_t)BATCH * 96 * HWPIX * 4);
  float*  uf     = (float*)take((size_t)BATCH * 96 * HWPIX * 4);
  float*  attL   = (float*)take((size_t)BATCH * 32 * HWPIX * 4);
  bf16_t* attT   = (bf16_t*)take((size_t)BATCH * HWPIX * 32 * 2);
  bf16_t* yT     = (bf16_t*)take((size_t)BATCH * HWPIX * 96 * 2);

  k_prep_x<<<128, 256, 0, stream>>>(x, xT);
  k_prep_w<<<302, 256, 0, stream>>>(w1a, w1c, w211, kbw, kbb, wproj,
                                    wcat, bfKW, bfBias, wProjB);
  k_mean<<<96, 256, 0, stream>>>(x, means);
  k_sca<<<1, 256, 0, stream>>>(means, wsca, bsca, sca);
  k_gemm1<<<256, 256, 0, stream>>>(xT, wcat, b1a, b1c, b211, hidA, hidC, attL);
  k_dw<<<12288, 256, 0, stream>>>(hidA, wdwa, bdwa, x1);
  k_dw<<<12288, 256, 0, stream>>>(hidC, wdwc, bdwc, uf);
  k_att<<<128, 256, 0, stream>>>(x, wc2a, bc2a, wc2b, bc2b, attg, attL, attT);
  k_kba<<<256, 256, 0, stream>>>(attT, bfKW, bfBias, uf, x1, sca, ga1, yT);
  k_proj<<<256, 256, 0, stream>>>(yT, wProjB, bproj, (float*)d_out);
}
